// SelfAttentionHead_89996744720418
// MI455X (gfx1250) — matrix operand order repaired and checked by reading, not compiled, not run on hardware
//
#include <hip/hip_runtime.h>
#include <hip/hip_bf16.h>
#include <math.h>

// ---------------------------------------------------------------------------
// Self-attention head (T=4096, model=2048, head=128) for MI455X / gfx1250.
//   kernel 0: one-shot f32->bf16 cast of x and Wq/Wk/Wv (coalesced, HBM
//             bound; 1/sqrt(128) folded into Wq before rounding). This keeps
//             all v_cvt out of the GEMM loops -> no WMMA->VALU WAR nops.
//   kernel 1: QKV projection, pure-bf16 v_wmma_f32_16x16x32_bf16. Each wave
//             owns a 16-row block; per k-chunk it preloads A + all 8 B
//             fragments (18x b128, clause-grouped) and then bursts 8 WMMAs,
//             so load latency overlaps XDL execution within the wave.
//   kernel 2: causal flash attention, 4 waves per 16-query block, each wave
//             processes key-pair-blocks strided by 4 with private online
//             softmax, merged with a log-sum-exp combine through LDS.
//             S is computed TRANSPOSED (K@Q^T) so the softmaxed P tile is
//             already in WMMA A-fragment layout -> zero-shuffle P@V.
// ---------------------------------------------------------------------------

#define T_TOK 4096
#define IN_SZ 2048
#define HEAD  128

typedef __attribute__((ext_vector_type(16))) __bf16 v16bf;
typedef __attribute__((ext_vector_type(8)))  __bf16 v8bf;
typedef __attribute__((ext_vector_type(8)))  float  v8f;

// ---- fragment loaders (bf16 sources only; hot loops have no converts) -----
// A-matrix 16x32 bf16 (ISA 7.12.2): lane<16 holds K = e (e<8) / e+8 (e>=8);
// lanes>=16 add +8.  hh = lane>>4.
static __device__ __forceinline__ v16bf frag_a_bf16(const __bf16* __restrict__ src,
                                                    int ld, int row, int k0, int hh) {
    const __bf16* p = src + (size_t)row * ld + k0 + hh * 8;
    v8bf lo = *(const v8bf*)p;          // 16B aligned
    v8bf hi = *(const v8bf*)(p + 16);
    return __builtin_shufflevector(lo, hi, 0,1,2,3,4,5,6,7,8,9,10,11,12,13,14,15);
}
// B-matrix 32x16 bf16: same K order as A (lane<16: K = e (e<8) / e+8 (e>=8);
// lanes>=16 add +8), column n = lane&15.
static __device__ __forceinline__ v16bf frag_b_bf16(const __bf16* __restrict__ src,
                                                    int ld, int col, int k0, int hh) {
    const __bf16* p = src + (size_t)col * ld + k0 + hh * 8;
    v8bf lo = *(const v8bf*)p;          // 16B aligned
    v8bf hi = *(const v8bf*)(p + 16);
    return __builtin_shufflevector(lo, hi, 0,1,2,3,4,5,6,7,8,9,10,11,12,13,14,15);
}

// ---- kernel 0: streaming f32 -> bf16 cast ---------------------------------
// grid = (4096, 4): y=0 -> x, y=1 -> Wq (scaled), y=2 -> Wk, y=3 -> Wv.
// 8 elements / thread: 2x b128 load, 1x b128 store.
__global__ __launch_bounds__(256) void cast_bf16_kernel(
    const float* __restrict__ x,  const float* __restrict__ Wq,
    const float* __restrict__ Wk, const float* __restrict__ Wv,
    __bf16* __restrict__ xb, __bf16* __restrict__ Wqb,
    __bf16* __restrict__ Wkb, __bf16* __restrict__ Wvb)
{
    const float* s; __bf16* d; int n; float sc;
    switch (blockIdx.y) {
        case 0:  s = x;  d = xb;  n = T_TOK * IN_SZ; sc = 1.0f; break;
        case 1:  s = Wq; d = Wqb; n = HEAD * IN_SZ;  sc = 0.08838834764831845f; break;
        case 2:  s = Wk; d = Wkb; n = HEAD * IN_SZ;  sc = 1.0f; break;
        default: s = Wv; d = Wvb; n = HEAD * IN_SZ;  sc = 1.0f; break;
    }
    const int idx = (blockIdx.x * 256 + threadIdx.x) * 8;
    if (idx >= n) return;
    const float4 f0 = *(const float4*)(s + idx);
    const float4 f1 = *(const float4*)(s + idx + 4);
    v8bf o;
    o[0] = (__bf16)(f0.x * sc); o[1] = (__bf16)(f0.y * sc);
    o[2] = (__bf16)(f0.z * sc); o[3] = (__bf16)(f0.w * sc);
    o[4] = (__bf16)(f1.x * sc); o[5] = (__bf16)(f1.y * sc);
    o[6] = (__bf16)(f1.z * sc); o[7] = (__bf16)(f1.w * sc);
    *(v8bf*)(d + idx) = o;
}

// ---- kernel 1: QKV projection (pure bf16) ---------------------------------
// grid = (T/128, 3), block = 256: 8 waves, each wave owns one 16-row block
// and computes all 8 column tiles, reusing its A fragment 8x per k-chunk.
__global__ __launch_bounds__(256) void qkv_proj_kernel(
    const __bf16* __restrict__ xb,  const __bf16* __restrict__ Wqb,
    const __bf16* __restrict__ Wkb, const __bf16* __restrict__ Wvb,
    __bf16* __restrict__ Qb, __bf16* __restrict__ Kb, __bf16* __restrict__ Vt)
{
    const int lane = threadIdx.x & 31;
    const int l15  = lane & 15;
    const int hh   = lane >> 4;
    const int wave = threadIdx.x >> 5;
    const int row0 = (blockIdx.x * 8 + wave) * 16;
    const int mat  = blockIdx.y;                 // 0=Q 1=K 2=V

    // uniform per-block configuration -> straight-line store loop
    const __bf16* W;  __bf16* dst;  int sr, sc;
    if (mat == 0)      { W = Wqb; dst = Qb; sr = HEAD; sc = 1; }
    else if (mat == 1) { W = Wkb; dst = Kb; sr = HEAD; sc = 1; }
    else               { W = Wvb; dst = Vt; sr = 1;    sc = T_TOK; }

    v8f acc[8] = {};
    for (int k0 = 0; k0 < IN_SZ; k0 += 32) {
        // Stage 1: issue ALL fragment loads (A + 8 B) before any WMMA, so
        // the waits can stagger (loadcnt 7..0) and latency overlaps XDL.
        const v16bf a = frag_a_bf16(xb, IN_SZ, row0 + l15, k0, hh);
        v16bf b[8];
#pragma unroll
        for (int t = 0; t < 8; ++t)
            b[t] = frag_b_bf16(W, IN_SZ, t * 16 + l15, k0, hh);
        // Stage 2: 8 back-to-back WMMAs on disjoint operands.
#pragma unroll
        for (int t = 0; t < 8; ++t)
            acc[t] = __builtin_amdgcn_wmma_f32_16x16x32_bf16(
                         false, a, false, b[t], (short)0, acc[t], false, false);
    }

#pragma unroll
    for (int t = 0; t < 8; ++t)
#pragma unroll
        for (int r = 0; r < 8; ++r) {
            const int m = row0 + r + hh * 8;             // C row
            const int n = t * 16 + l15;                  // C col
            dst[(size_t)m * sr + (size_t)n * sc] = (__bf16)acc[t][r];
        }
}

// ---- kernel 2: causal flash attention ------------------------------------
// grid = T/16 blocks (reverse-scheduled), block = 128 (4 waves).
// Wave w handles key-pair-blocks jp = w, w+4, ... with private online
// softmax; results merged via log-sum-exp combine in LDS.
__global__ __launch_bounds__(128) void flash_attn_kernel(
    const __bf16* __restrict__ Qb, const __bf16* __restrict__ Kb,
    const __bf16* __restrict__ Vt, float* __restrict__ out)
{
    __shared__ float sO[4][16][HEAD];   // 32 KB (of 320 KB/WGP)
    __shared__ float sM[4][16];
    __shared__ float sL[4][16];

    const int lane  = threadIdx.x & 31;
    const int l15   = lane & 15;
    const int hh    = lane >> 4;
    const int wave  = threadIdx.x >> 5;
    const int iblk  = gridDim.x - 1 - blockIdx.x;   // heaviest blocks first
    const int qbase = iblk * 16;
    const int qglob = qbase + l15;                  // this lane's query row

    // Q as B-fragments (col = query), loaded once per wave
    v16bf bq[4];
#pragma unroll
    for (int c = 0; c < 4; ++c)
        bq[c] = frag_b_bf16(Qb, HEAD, qbase + l15, c * 32, hh);

    v8f o[8] = {};                                  // O accum: 8 head-dim tiles
    float run_m = -1e30f, run_l = 0.0f;

    const int npair = (iblk >> 1) + 1;              // 32-key pair blocks needed
    for (int jp = wave; jp < npair; jp += 4) {
        const int kb = jp * 32;

        // S^T = K @ Q^T (row = key, col = query) for two 16-key blocks
        v8f s1 = {}, s2 = {};
#pragma unroll
        for (int c = 0; c < 4; ++c) {
            v16bf ak1 = frag_a_bf16(Kb, HEAD, kb + l15,      c * 32, hh);
            v16bf ak2 = frag_a_bf16(Kb, HEAD, kb + 16 + l15, c * 32, hh);
            s1 = __builtin_amdgcn_wmma_f32_16x16x32_bf16(
                     false, ak1, false, bq[c], (short)0, s1, false, false);
            s2 = __builtin_amdgcn_wmma_f32_16x16x32_bf16(
                     false, ak2, false, bq[c], (short)0, s2, false, false);
        }

        // causal mask (only pairs straddling the diagonal)
        if (kb + 31 > qbase) {
#pragma unroll
            for (int r = 0; r < 8; ++r) {
                const int key = kb + r + hh * 8;    // S^T row index
                if (key      > qglob) s1[r] = -1e30f;
                if (key + 16 > qglob) s2[r] = -1e30f;
            }
        }

        // online softmax over 32 keys; query m = lane&15, halves via xor-16
        float mx = run_m;
#pragma unroll
        for (int r = 0; r < 8; ++r) mx = fmaxf(mx, fmaxf(s1[r], s2[r]));
        mx = fmaxf(mx, __shfl_xor(mx, 16, 32));

        const float alpha = __expf(run_m - mx);
        float p1[8], p2[8], lsum = 0.0f;
#pragma unroll
        for (int r = 0; r < 8; ++r) {
            p1[r] = __expf(s1[r] - mx);
            p2[r] = __expf(s2[r] - mx);
            lsum += p1[r] + p2[r];
        }
        lsum += __shfl_xor(lsum, 16, 32);
        run_l = run_l * alpha + lsum;
        run_m = mx;

        // P is already in A-fragment layout (S was computed transposed)
        v16bf ap;
#pragma unroll
        for (int e = 0; e < 8; ++e) { ap[e] = (__bf16)p1[e]; ap[e + 8] = (__bf16)p2[e]; }

        // per-row alpha rescale of O, then O += P @ V
        float aRow[8];
#pragma unroll
        for (int r = 0; r < 8; ++r) aRow[r] = __shfl(alpha, r + hh * 24, 32);
#pragma unroll
        for (int t = 0; t < 8; ++t) {
            v8f acc = o[t];
#pragma unroll
            for (int r = 0; r < 8; ++r) acc[r] *= aRow[r];
            // B-fragment of V: B[k][n] = V[kb+k][t*16+n] = Vt[t*16+n][kb+k]
            v16bf bv = frag_b_bf16(Vt, T_TOK, t * 16 + l15, kb, hh);
            o[t] = __builtin_amdgcn_wmma_f32_16x16x32_bf16(
                       false, ap, false, bv, (short)0, acc, false, false);
        }
    }

    // ---- merge the 4 partial softmaxes through LDS -----------------------
    if (hh == 0) { sM[wave][l15] = run_m; sL[wave][l15] = run_l; }
#pragma unroll
    for (int t = 0; t < 8; ++t)
#pragma unroll
        for (int r = 0; r < 8; ++r)
            sO[wave][r + hh * 8][t * 16 + l15] = o[t][r];
    __syncthreads();

    // 128 threads x 16 outputs: thread -> (row, 16-col span)
    const int row = threadIdx.x >> 3;
    const int c0  = (threadIdx.x & 7) * 16;
    float mstar = sM[0][row];
#pragma unroll
    for (int w = 1; w < 4; ++w) mstar = fmaxf(mstar, sM[w][row]);

    float num[16] = {};
    float den = 0.0f;
#pragma unroll
    for (int w = 0; w < 4; ++w) {
        const float ww = __expf(sM[w][row] - mstar);
        den += ww * sL[w][row];
#pragma unroll
        for (int e = 0; e < 16; ++e) num[e] += ww * sO[w][row][c0 + e];
    }
    const float inv = 1.0f / den;
#pragma unroll
    for (int e = 0; e < 16; ++e)
        out[(size_t)(qbase + row) * HEAD + c0 + e] = num[e] * inv;
}

// ---- host launch ----------------------------------------------------------
extern "C" void kernel_launch(void* const* d_in, const int* in_sizes, int n_in,
                              void* d_out, int out_size, void* d_ws, size_t ws_size,
                              hipStream_t stream) {
    (void)in_sizes; (void)n_in; (void)out_size; (void)ws_size;
    const float* x  = (const float*)d_in[0];
    const float* Wq = (const float*)d_in[1];
    const float* Wk = (const float*)d_in[2];
    const float* Wv = (const float*)d_in[3];

    __bf16* Qb  = (__bf16*)d_ws;                      // [T, HEAD]    1 MB
    __bf16* Kb  = Qb  + (size_t)T_TOK * HEAD;         // [T, HEAD]    1 MB
    __bf16* Vt  = Kb  + (size_t)T_TOK * HEAD;         // [HEAD, T]    1 MB
    __bf16* xb  = Vt  + (size_t)T_TOK * HEAD;         // [T, IN]     16 MB
    __bf16* Wqb = xb  + (size_t)T_TOK * IN_SZ;        // [HEAD, IN] 0.5 MB
    __bf16* Wkb = Wqb + (size_t)HEAD * IN_SZ;
    __bf16* Wvb = Wkb + (size_t)HEAD * IN_SZ;

    dim3 g0(T_TOK * IN_SZ / (256 * 8), 4);            // (4096, 4)
    cast_bf16_kernel<<<g0, 256, 0, stream>>>(x, Wq, Wk, Wv, xb, Wqb, Wkb, Wvb);

    dim3 g1(T_TOK / 128, 3);
    qkv_proj_kernel<<<g1, 256, 0, stream>>>(xb, Wqb, Wkb, Wvb, Qb, Kb, Vt);

    flash_attn_kernel<<<T_TOK / 16, 128, 0, stream>>>(Qb, Kb, Vt, (float*)d_out);
}
